// Encoder_L1_88098369176150
// MI455X (gfx1250) — matrix operand order repaired and checked by reading, not compiled, not run on hardware
//
#include <hip/hip_runtime.h>
#include <hip/hip_bf16.h>
#include <cstdint>

// ---------------------------------------------------------------------------
// GNN encoder: GIN x2 (+BN+LeakyReLU) -> GAT x2, output [N, 4*64] concat.
// GEMMs use CDNA5 v_wmma_f32_16x16x32_f16 (wave32), weights pre-packed in
// fragment order so all LDS fragment traffic is ds_load_b128.
// ---------------------------------------------------------------------------

typedef _Float16 v16h __attribute__((ext_vector_type(16)));
typedef _Float16 v8h  __attribute__((ext_vector_type(8)));
typedef _Float16 v4h  __attribute__((ext_vector_type(4)));
typedef float    v8f  __attribute__((ext_vector_type(8)));
typedef float    v4f  __attribute__((ext_vector_type(4)));

#define D_IN 128
#define D_H  64
#define NHEAD 4

__device__ __forceinline__ float leaky(float x, float s) { return x >= 0.f ? x : s * x; }

// ordered-uint encoding of float for atomicMax (monotone bijection)
__device__ __forceinline__ unsigned f2o(float f) {
    unsigned u = __float_as_uint(f);
    return (u & 0x80000000u) ? ~u : (u | 0x80000000u);
}
__device__ __forceinline__ float o2f(unsigned u) {
    return (u & 0x80000000u) ? __uint_as_float(u & 0x7fffffffu) : __uint_as_float(~u);
}

__device__ __forceinline__ v8f wmma16(v16h a, v16h b, v8f c) {
    // D = A(16x32 f16) * B(32x16 f16) + C(16x16 f32)
    return __builtin_amdgcn_wmma_f32_16x16x32_f16(false, a, false, b, (short)0, c, false, false);
}

#define CAT16(lo, hi) __builtin_shufflevector(lo, hi, 0,1,2,3,4,5,6,7,8,9,10,11,12,13,14,15)

// A fragment from row-major 16xK f16 tile in LDS (leading dim ld), k-step kt.
// Lane l<16: row M=l, elems 0..7 -> K 0..7, 8..15 -> K 16..23.
// Lane l>=16: row M=l-16, elems 0..7 -> K 8..15, 8..15 -> K 24..31.
__device__ __forceinline__ v16h load_a_frag(const _Float16* tile, int ld, int kt) {
    const int lane = threadIdx.x & 31;
    const int m = lane & 15;
    const int koff = (lane < 16) ? 0 : 8;
    const _Float16* p = tile + m * ld + kt * 32 + koff;
    v8h lo = *(const v8h*)p;
    v8h hi = *(const v8h*)(p + 16);
    return CAT16(lo, hi);
}

// B fragment from pre-packed weights: layout [(kt*NT+nt)][lane][16 halves],
// i.e. each lane reads 16 contiguous halves (2x ds_load_b128).
__device__ __forceinline__ v16h load_b_packed(const _Float16* wp, int kt, int nt, int NT) {
    const int lane = threadIdx.x & 31;
    const _Float16* p = wp + (((kt * NT + nt) << 5) + lane) * 16;
    v8h lo = *(const v8h*)p;
    v8h hi = *(const v8h*)(p + 8);
    return CAT16(lo, hi);
}

// ---------------------------------------------------------------------------
// Utility kernels
// ---------------------------------------------------------------------------
__global__ void fill_f32_kernel(float* __restrict__ p, float v, long long cnt) {
    long long i = (long long)blockIdx.x * blockDim.x + threadIdx.x;
    if (i < cnt) p[i] = v;
}

// fetch source element of packed index p for a row-major KxN weight (NT = N/16)
__device__ __forceinline__ float pack_src(const float* __restrict__ w, int N, int NT, int p) {
    int e = p & 15;
    int l = (p >> 4) & 31;
    int tile = p >> 9;
    int nt = tile % NT, kt = tile / NT;
    int k = kt * 32 + ((l < 16) ? 0 : 8) + (e & 7) + ((e & 8) ? 16 : 0);
    int nn = nt * 16 + (l & 15);
    return w[k * N + nn];
}

// convert + fragment-pack all weight matrices:
// [w1_0 16384 | w2_0 8192 | w1_1 4096 | w2_1 4096 | gat0 16384 | gat1 16384]
__global__ void prep_weights_kernel(const float* __restrict__ w10, const float* __restrict__ w20,
                                    const float* __restrict__ w11, const float* __restrict__ w21,
                                    const float* __restrict__ g0, const float* __restrict__ g1,
                                    _Float16* __restrict__ out) {
    int i = blockIdx.x * blockDim.x + threadIdx.x;
    if (i >= 65536) return;
    float v;
    if (i < 16384)       v = pack_src(w10, 128, 8, i);
    else if (i < 24576)  v = pack_src(w20, 64, 4, i - 16384);
    else if (i < 28672)  v = pack_src(w11, 64, 4, i - 24576);
    else if (i < 32768)  v = pack_src(w21, 64, 4, i - 28672);
    else if (i < 49152) { int j = i - 32768; v = pack_src(g0 + (j >> 12) * 4096, 64, 4, j & 4095); }
    else                { int j = i - 49152; v = pack_src(g1 + (j >> 12) * 4096, 64, 4, j & 4095); }
    out[i] = (_Float16)v;
}

// scatter-add of D features per edge (D % 4 == 0); one thread = 4 channels
__global__ void scatter_add_kernel(const float* __restrict__ x, int xstride, int xcoloff,
                                   const long long* __restrict__ src, const long long* __restrict__ dst,
                                   float* __restrict__ agg, int aggstride, int E, int Dq) {
    int idx = blockIdx.x * blockDim.x + threadIdx.x;
    if (idx >= E * Dq) return;
    int e = idx / Dq, c4 = (idx - e * Dq) * 4;
    int s = (int)src[e], d = (int)dst[e];
    const float* xp = x + (size_t)s * xstride + xcoloff + c4;
    float* ap = agg + (size_t)d * aggstride + c4;
    atomicAdd(ap + 0, xp[0]);
    atomicAdd(ap + 1, xp[1]);
    atomicAdd(ap + 2, xp[2]);
    atomicAdd(ap + 3, xp[3]);
}

// ---------------------------------------------------------------------------
// GIN MLP:  out = (leaky((1+eps)x + agg) @ W1 + b1, 0.01) @ W2 + b2
// block = 256 threads = 8 waves; each wave owns a 16-row tile.
// ---------------------------------------------------------------------------
template <int DIN, int DMID, int DOUT>
__global__ __launch_bounds__(256) void gin_mlp_kernel(
        const float* __restrict__ xsrc, int xstride, int xcoloff,
        const float* __restrict__ agg,
        const _Float16* __restrict__ w1, const _Float16* __restrict__ w2,
        const float* __restrict__ b1, const float* __restrict__ b2,
        const float* __restrict__ eps_p,
        float* __restrict__ out, int n) {
    __shared__ __align__(16) _Float16 sW1[DIN * DMID];
    __shared__ __align__(16) _Float16 sW2[DMID * DOUT];
    __shared__ __align__(16) _Float16 sH[8][16 * DIN];
    __shared__ __align__(16) _Float16 sC1[8][16 * DMID];
    __shared__ __align__(16) float    sO[8][DOUT * 16];   // column-major [col][row]

    const int tid = threadIdx.x, wave = tid >> 5, lane = tid & 31;
    for (int i = tid; i < DIN * DMID / 8; i += 256) ((v8h*)sW1)[i] = ((const v8h*)w1)[i];
    for (int i = tid; i < DMID * DOUT / 8; i += 256) ((v8h*)sW2)[i] = ((const v8h*)w2)[i];

    const float eps1 = 1.0f + eps_p[0];
    const int row0 = (blockIdx.x * 8 + wave) * 16;
    const bool full = (row0 + 16 <= n);

    // h = (1+eps)*x + agg, staged to LDS as f16 (vectorized)
    for (int i = lane; i < 16 * (DIN / 4); i += 32) {
        int r = i / (DIN / 4), c4 = (i % (DIN / 4)) * 4;
        int gr = row0 + r;
        v4h hv;
        if (full || gr < n) {
            v4f xv = *(const v4f*)(xsrc + (size_t)gr * xstride + xcoloff + c4);
            v4f av = *(const v4f*)(agg + (size_t)gr * DIN + c4);
#pragma unroll
            for (int j = 0; j < 4; ++j) hv[j] = (_Float16)(eps1 * xv[j] + av[j]);
        } else {
            hv = (v4h)(_Float16)0.f;
        }
        *(v4h*)(&sH[wave][r * DIN + c4]) = hv;
    }
    __syncthreads();

    // GEMM1: [16 x DIN] * [DIN x DMID] + b1 -> leaky -> sC1 (f16)
#pragma unroll
    for (int nt = 0; nt < DMID / 16; ++nt) {
        v8f acc = {};
#pragma unroll
        for (int kt = 0; kt < DIN / 32; ++kt) {
            v16h a = load_a_frag(&sH[wave][0], DIN, kt);
            v16h b = load_b_packed(sW1, kt, nt, DMID / 16);
            acc = wmma16(a, b, acc);
        }
        const int nn = (lane & 15) + nt * 16;
        const int mbase = (lane < 16) ? 0 : 8;
        const float bv = b1[nn];
#pragma unroll
        for (int v = 0; v < 8; ++v)
            sC1[wave][(mbase + v) * DMID + nn] = (_Float16)leaky(acc[v] + bv, 0.01f);
    }
    __syncthreads();

    // GEMM2: [16 x DMID] * [DMID x DOUT] + b2 -> sO (column-major, vector stores)
#pragma unroll
    for (int nt = 0; nt < DOUT / 16; ++nt) {
        v8f acc = {};
#pragma unroll
        for (int kt = 0; kt < DMID / 32; ++kt) {
            v16h a = load_a_frag(&sC1[wave][0], DMID, kt);
            v16h b = load_b_packed(sW2, kt, nt, DOUT / 16);
            acc = wmma16(a, b, acc);
        }
        const int nn = (lane & 15) + nt * 16;
        const int mbase = (lane < 16) ? 0 : 8;
        const float bv = b2[nn];
        float* q = &sO[wave][nn * 16 + mbase];
        v4f lo, hi;
#pragma unroll
        for (int v = 0; v < 4; ++v) { lo[v] = acc[v] + bv; hi[v] = acc[4 + v] + bv; }
        *(v4f*)q = lo;
        *(v4f*)(q + 4) = hi;
    }
    __syncthreads();

    // coalesced row-major writeback (global_store_b128)
    for (int i = lane; i < 16 * (DOUT / 4); i += 32) {
        int q = i % (DOUT / 4), r = i / (DOUT / 4);
        int gr = row0 + r;
        if (full || gr < n) {
            v4f val;
#pragma unroll
            for (int j = 0; j < 4; ++j) val[j] = sO[wave][(q * 4 + j) * 16 + r];
            *(v4f*)(out + (size_t)gr * DOUT + q * 4) = val;
        }
    }
}

// ---------------------------------------------------------------------------
// BatchNorm (training-mode biased stats) + LeakyReLU
// ---------------------------------------------------------------------------
__global__ void bn_stats_kernel(const float* __restrict__ x, int n, float* __restrict__ stats) {
    const int c = threadIdx.x;            // blockDim = 64
    const int r0 = blockIdx.x * 512;
    float s = 0.f, ss = 0.f;
    for (int i = 0; i < 512; ++i) {
        int r = r0 + i;
        if (r < n) {
            float v = x[(size_t)r * 64 + c];
            s += v; ss += v * v;
        }
    }
    atomicAdd(&stats[c], s);
    atomicAdd(&stats[64 + c], ss);
}

__global__ void bn_apply_kernel(const float* __restrict__ x, const float* __restrict__ stats,
                                const float* __restrict__ g, const float* __restrict__ b,
                                float* __restrict__ out, int outstride, int coloff, int n) {
    int idx = blockIdx.x * blockDim.x + threadIdx.x;
    if (idx >= n * 64) return;
    int r = idx >> 6, c = idx & 63;
    float invn = 1.0f / (float)n;
    float mu = stats[c] * invn;
    float var = stats[64 + c] * invn - mu * mu;
    float y = (x[(size_t)r * 64 + c] - mu) * rsqrtf(var + 1e-5f) * g[c] + b[c];
    out[(size_t)r * outstride + coloff + c] = leaky(y, 0.01f);
}

// ---------------------------------------------------------------------------
// GAT: per-head projection xe[n,h,c] = x @ w[h]  (WMMA f16 -> f16 output)
// ---------------------------------------------------------------------------
__global__ __launch_bounds__(256) void gat_xe_kernel(
        const float* __restrict__ xbase, int xstride, int xcoloff,
        const _Float16* __restrict__ w,   // packed [4 heads][KT=2][NT=4][32][16]
        _Float16* __restrict__ xe, int n) {
    __shared__ __align__(16) _Float16 sW[NHEAD * 64 * 64];
    __shared__ __align__(16) _Float16 sX[8][16 * 64];
    __shared__ __align__(16) _Float16 sOx[8][16 * 256];   // row-major staged output

    const int tid = threadIdx.x, wave = tid >> 5, lane = tid & 31;
    for (int i = tid; i < NHEAD * 64 * 64 / 8; i += 256) ((v8h*)sW)[i] = ((const v8h*)w)[i];

    const int row0 = (blockIdx.x * 8 + wave) * 16;
    const bool full = (row0 + 16 <= n);

    for (int i = lane; i < 16 * 16; i += 32) {
        int r = i >> 4, c4 = (i & 15) * 4;
        int gr = row0 + r;
        v4h hv;
        if (full || gr < n) {
            v4f xv = *(const v4f*)(xbase + (size_t)gr * xstride + xcoloff + c4);
#pragma unroll
            for (int j = 0; j < 4; ++j) hv[j] = (_Float16)xv[j];
        } else {
            hv = (v4h)(_Float16)0.f;
        }
        *(v4h*)(&sX[wave][r * 64 + c4]) = hv;
    }
    __syncthreads();

#pragma unroll
    for (int h = 0; h < NHEAD; ++h) {
#pragma unroll
        for (int nt = 0; nt < 4; ++nt) {
            v8f acc = {};
#pragma unroll
            for (int kt = 0; kt < 2; ++kt) {
                v16h a = load_a_frag(&sX[wave][0], 64, kt);
                v16h b = load_b_packed(&sW[h * 4096], kt, nt, 4);
                acc = wmma16(a, b, acc);
            }
            const int nn = (lane & 15) + nt * 16;
            const int mbase = (lane < 16) ? 0 : 8;
#pragma unroll
            for (int v = 0; v < 8; ++v)
                sOx[wave][(mbase + v) * 256 + h * 64 + nn] = (_Float16)acc[v];
        }
    }
    __syncthreads();

    // coalesced f16 writeback: 16 x global_store_b128 per wave
    for (int i = lane; i < 512; i += 32) {
        int q = i & 31, r = i >> 5;
        int gr = row0 + r;
        if (full || gr < n) {
            v8h val = *(const v8h*)(&sOx[wave][r * 256 + q * 8]);
            *(v8h*)(xe + (size_t)gr * 256 + q * 8) = val;
        }
    }
}

// attention logits per (node, head)
__global__ void gat_al_kernel(const _Float16* __restrict__ xe,
                              const float* __restrict__ asrc, const float* __restrict__ adst,
                              float* __restrict__ als, float* __restrict__ ald, int n) {
    int idx = blockIdx.x * blockDim.x + threadIdx.x;
    if (idx >= n * NHEAD) return;
    int nn = idx >> 2, h = idx & 3;
    const v8h* xp = (const v8h*)(xe + (size_t)nn * 256 + h * 64);
    const v4f* as = (const v4f*)(asrc + h * 64);
    const v4f* ad = (const v4f*)(adst + h * 64);
    float s = 0.f, d = 0.f;
#pragma unroll
    for (int c8 = 0; c8 < 8; ++c8) {
        v8h xv = xp[c8];
        v4f a0 = as[c8 * 2], a1 = as[c8 * 2 + 1];
        v4f d0 = ad[c8 * 2], d1 = ad[c8 * 2 + 1];
#pragma unroll
        for (int j = 0; j < 4; ++j) {
            float lo = (float)xv[j], hi = (float)xv[4 + j];
            s += lo * a0[j] + hi * a1[j];
            d += lo * d0[j] + hi * d1[j];
        }
    }
    als[idx] = s;
    ald[idx] = d;
}

__device__ __forceinline__ void edge_sd(const long long* src, const long long* dst,
                                        int e, int E, int& s, int& d) {
    if (e < E) { s = (int)src[e]; d = (int)dst[e]; }
    else       { s = d = e - E; }   // self loops appended
}

__global__ void edge_max_kernel(const long long* __restrict__ src, const long long* __restrict__ dst,
                                int E, int EN,
                                const float* __restrict__ als, const float* __restrict__ ald,
                                unsigned* __restrict__ emax) {
    int e = blockIdx.x * blockDim.x + threadIdx.x;
    if (e >= EN) return;
    int s, d;
    edge_sd(src, dst, e, E, s, d);
#pragma unroll
    for (int h = 0; h < NHEAD; ++h) {
        float v = leaky(als[s * NHEAD + h] + ald[d * NHEAD + h], 0.2f);
        atomicMax(&emax[d * NHEAD + h], f2o(v));
    }
}

__global__ void edge_expsum_kernel(const long long* __restrict__ src, const long long* __restrict__ dst,
                                   int E, int EN,
                                   const float* __restrict__ als, const float* __restrict__ ald,
                                   const unsigned* __restrict__ emax, float* __restrict__ den) {
    int e = blockIdx.x * blockDim.x + threadIdx.x;
    if (e >= EN) return;
    int s, d;
    edge_sd(src, dst, e, E, s, d);
#pragma unroll
    for (int h = 0; h < NHEAD; ++h) {
        float v = leaky(als[s * NHEAD + h] + ald[d * NHEAD + h], 0.2f);
        float m = o2f(emax[d * NHEAD + h]);
        atomicAdd(&den[d * NHEAD + h], __expf(v - m));
    }
}

__global__ void edge_coef_kernel(const long long* __restrict__ src, const long long* __restrict__ dst,
                                 int E, int EN,
                                 const float* __restrict__ als, const float* __restrict__ ald,
                                 const unsigned* __restrict__ emax, const float* __restrict__ den,
                                 float* __restrict__ coef) {
    int idx = blockIdx.x * blockDim.x + threadIdx.x;
    if (idx >= EN * NHEAD) return;
    int e = idx >> 2, h = idx & 3;
    int s, d;
    edge_sd(src, dst, e, E, s, d);
    float v = leaky(als[s * NHEAD + h] + ald[d * NHEAD + h], 0.2f);
    float m = o2f(emax[d * NHEAD + h]);
    coef[idx] = __expf(v - m) / (den[d * NHEAD + h] + 1e-16f);
}

// weighted scatter: accum[d,h,c] += coef[e,h] * xe[s,h,c]; one thread = one (edge, c)
__global__ void edge_agg_kernel(const long long* __restrict__ src, const long long* __restrict__ dst,
                                int E, int EN,
                                const _Float16* __restrict__ xe, const float* __restrict__ coef,
                                float* __restrict__ accum) {
    int idx = blockIdx.x * blockDim.x + threadIdx.x;
    if (idx >= EN * 64) return;
    int e = idx >> 6, c = idx & 63;
    int s, d;
    edge_sd(src, dst, e, E, s, d);
    const _Float16* xp = xe + (size_t)s * 256 + c;
    float* ap = accum + (size_t)d * 256 + c;
#pragma unroll
    for (int h = 0; h < NHEAD; ++h)
        atomicAdd(ap + h * 64, coef[e * NHEAD + h] * (float)xp[h * 64]);
}

__global__ void gat_out_kernel(const float* __restrict__ accum, const float* __restrict__ bias,
                               float* __restrict__ out, int outstride, int coloff, int n) {
    int idx = blockIdx.x * blockDim.x + threadIdx.x;
    if (idx >= n * 64) return;
    int r = idx >> 6, c = idx & 63;
    const float* ap = accum + (size_t)r * 256 + c;
    float v = 0.25f * (ap[0] + ap[64] + ap[128] + ap[192]) + bias[c];
    out[(size_t)r * outstride + coloff + c] = leaky(v, 0.01f);
}

// ---------------------------------------------------------------------------
// Host-side orchestration
// ---------------------------------------------------------------------------
#define CDIV(a, b) (((a) + (b) - 1) / (b))

extern "C" void kernel_launch(void* const* d_in, const int* in_sizes, int n_in,
                              void* d_out, int out_size, void* d_ws, size_t ws_size,
                              hipStream_t stream) {
    const int n = in_sizes[0] / D_IN;
    const int E = in_sizes[1] / 2;
    const int EN = E + n;

    const float* feat = (const float*)d_in[0];
    const long long* ei = (const long long*)d_in[1];
    const long long* esrc = ei;
    const long long* edst = ei + E;
    const float* gin0_w1 = (const float*)d_in[2];
    const float* gin0_b1 = (const float*)d_in[3];
    const float* gin0_w2 = (const float*)d_in[4];
    const float* gin0_b2 = (const float*)d_in[5];
    const float* gin0_eps = (const float*)d_in[6];
    const float* bn0_g = (const float*)d_in[7];
    const float* bn0_b = (const float*)d_in[8];
    const float* gin1_w1 = (const float*)d_in[9];
    const float* gin1_b1 = (const float*)d_in[10];
    const float* gin1_w2 = (const float*)d_in[11];
    const float* gin1_b2 = (const float*)d_in[12];
    const float* gin1_eps = (const float*)d_in[13];
    const float* bn1_g = (const float*)d_in[14];
    const float* bn1_b = (const float*)d_in[15];
    const float* gat0_asrc = (const float*)d_in[17];
    const float* gat0_adst = (const float*)d_in[18];
    const float* gat0_bias = (const float*)d_in[19];
    const float* gat1_asrc = (const float*)d_in[21];
    const float* gat1_adst = (const float*)d_in[22];
    const float* gat1_bias = (const float*)d_in[23];
    float* out = (float*)d_out;

    // workspace layout
    char* ws = (char*)d_ws;
    size_t offA = 0;                                      // N*256 f32: GIN agg(+xpre) / GAT accum
    size_t offXE = offA + (size_t)n * 256 * 4;            // N*256 f16: xe
    size_t offALS = offXE + (size_t)n * 256 * 2;
    size_t offALD = offALS + (size_t)n * NHEAD * 4;
    size_t offEMX = offALD + (size_t)n * NHEAD * 4;
    size_t offDEN = offEMX + (size_t)n * NHEAD * 4;
    size_t offCOEF = offDEN + (size_t)n * NHEAD * 4;
    size_t offSTATS = offCOEF + (size_t)EN * NHEAD * 4;
    size_t offWH = offSTATS + 1024;

    float* A = (float*)(ws + offA);
    float* xpre = A + (size_t)n * 128;
    _Float16* XE = (_Float16*)(ws + offXE);
    float* ALS = (float*)(ws + offALS);
    float* ALD = (float*)(ws + offALD);
    unsigned* EMX = (unsigned*)(ws + offEMX);
    float* DEN = (float*)(ws + offDEN);
    float* COEF = (float*)(ws + offCOEF);
    float* STATS = (float*)(ws + offSTATS);
    _Float16* WH = (_Float16*)(ws + offWH);
    const _Float16* w1_0h = WH;
    const _Float16* w2_0h = WH + 16384;
    const _Float16* w1_1h = WH + 24576;
    const _Float16* w2_1h = WH + 28672;
    const _Float16* gat0h = WH + 32768;
    const _Float16* gat1h = WH + 49152;

    const int T = 256;
    const int mlpBlocks = CDIV(n, 128);   // 8 waves x 16 rows per block

    // weight conversion + fragment packing
    prep_weights_kernel<<<CDIV(65536, T), T, 0, stream>>>(
        gin0_w1, gin0_w2, gin1_w1, gin1_w2,
        (const float*)d_in[16], (const float*)d_in[20], WH);

    // ---------------- GIN 0 ----------------
    fill_f32_kernel<<<CDIV((long long)n * 128, T), T, 0, stream>>>(A, 0.f, (long long)n * 128);
    scatter_add_kernel<<<CDIV(E * 32, T), T, 0, stream>>>(feat, D_IN, 0, esrc, edst, A, D_IN, E, 32);
    gin_mlp_kernel<128, 128, 64><<<mlpBlocks, T, 0, stream>>>(
        feat, D_IN, 0, A, w1_0h, w2_0h, gin0_b1, gin0_b2, gin0_eps, xpre, n);
    fill_f32_kernel<<<1, 128, 0, stream>>>(STATS, 0.f, 128);
    bn_stats_kernel<<<CDIV(n, 512), 64, 0, stream>>>(xpre, n, STATS);
    bn_apply_kernel<<<CDIV(n * 64, T), T, 0, stream>>>(xpre, STATS, bn0_g, bn0_b, out, 256, 0, n);

    // ---------------- GIN 1 ----------------
    fill_f32_kernel<<<CDIV((long long)n * 64, T), T, 0, stream>>>(A, 0.f, (long long)n * 64);
    scatter_add_kernel<<<CDIV(E * 16, T), T, 0, stream>>>(out, 256, 0, esrc, edst, A, D_H, E, 16);
    gin_mlp_kernel<64, 64, 64><<<mlpBlocks, T, 0, stream>>>(
        out, 256, 0, A, w1_1h, w2_1h, gin1_b1, gin1_b2, gin1_eps, xpre, n);
    fill_f32_kernel<<<1, 128, 0, stream>>>(STATS, 0.f, 128);
    bn_stats_kernel<<<CDIV(n, 512), 64, 0, stream>>>(xpre, n, STATS);
    bn_apply_kernel<<<CDIV(n * 64, T), T, 0, stream>>>(xpre, STATS, bn1_g, bn1_b, out, 256, 64, n);

    // ---------------- GAT layers ----------------
    for (int layer = 0; layer < 2; ++layer) {
        const int xcol = (layer == 0) ? 64 : 128;
        const int ocol = (layer == 0) ? 128 : 192;
        const _Float16* wh = (layer == 0) ? gat0h : gat1h;
        const float* asrc = (layer == 0) ? gat0_asrc : gat1_asrc;
        const float* adst = (layer == 0) ? gat0_adst : gat1_adst;
        const float* bias = (layer == 0) ? gat0_bias : gat1_bias;

        gat_xe_kernel<<<mlpBlocks, T, 0, stream>>>(out, 256, xcol, wh, XE, n);
        gat_al_kernel<<<CDIV(n * NHEAD, T), T, 0, stream>>>(XE, asrc, adst, ALS, ALD, n);
        fill_f32_kernel<<<CDIV((long long)n * NHEAD, T), T, 0, stream>>>((float*)EMX, 0.f, (long long)n * NHEAD);
        fill_f32_kernel<<<CDIV((long long)n * NHEAD, T), T, 0, stream>>>(DEN, 0.f, (long long)n * NHEAD);
        fill_f32_kernel<<<CDIV((long long)n * 256, T), T, 0, stream>>>(A, 0.f, (long long)n * 256);
        edge_max_kernel<<<CDIV(EN, T), T, 0, stream>>>(esrc, edst, E, EN, ALS, ALD, EMX);
        edge_expsum_kernel<<<CDIV(EN, T), T, 0, stream>>>(esrc, edst, E, EN, ALS, ALD, EMX, DEN);
        edge_coef_kernel<<<CDIV(EN * NHEAD, T), T, 0, stream>>>(esrc, edst, E, EN, ALS, ALD, EMX, DEN, COEF);
        edge_agg_kernel<<<CDIV((long long)EN * 64, T), T, 0, stream>>>(esrc, edst, E, EN, XE, COEF, A);
        gat_out_kernel<<<CDIV(n * 64, T), T, 0, stream>>>(A, bias, out, 256, ocol, n);
    }

    (void)n_in; (void)out_size; (void)ws_size;
}
